// BaseRNN_57964878627417
// MI455X (gfx1250) — matrix operand order repaired and checked by reading, not compiled, not run on hardware
//
#include <hip/hip_runtime.h>
#include <math.h>

// ---------------------------------------------------------------------------
// SimpleRNN forward for MI455X (gfx1250, wave32, WMMA f16->f32).
//   logits[b,t,:] = (tanh-scan hidden states) @ Wd + bd
// Phases:
//   A) convert/transpose weights to f16 (N-major so WMMA B-frags are contiguous)
//   B) xp = gather(emb)[B*T,256] @ Wx -> f16, stored time-major [T][B][U]
//   C) sequential scan, 1 workgroup of 16 waves; h in LDS (XOR-swizzled);
//      half of Wh register-resident (loop-invariant), half streamed from L2
//   D) logits = hs @ Wd + bd -> f32 out
// ---------------------------------------------------------------------------

typedef _Float16 f16;
typedef __attribute__((ext_vector_type(16))) _Float16 v16h;
typedef __attribute__((ext_vector_type(8)))  _Float16 v8h;
typedef __attribute__((ext_vector_type(8)))  float    v8f;

#define VOCABN 256
#define UNITS  512
#define BATCH  64
#define TLEN   1024

// workspace layout, in f16 elements
#define EMB_OFF 0u                       // [256][256]
#define WXT_OFF 65536u                   // [512][256]  (WxT: N-major)
#define WHT_OFF 196608u                  // [512][512]  (WhT: N-major)
#define WDT_OFF 458752u                  // [256][512]  (WdT: N-major)
#define XP_OFF  589824u                  // [T][B][U]
#define HS_OFF  34144256u                // [T][B][U]

// ---------------------------------------------------------------------------
// Phase A: f32 -> f16 conversion + transposition of weights
// ---------------------------------------------------------------------------
__global__ void __launch_bounds__(256)
convert_weights(const float* __restrict__ emb, const float* __restrict__ Wx,
                const float* __restrict__ Wh, const float* __restrict__ Wd,
                f16* __restrict__ ws) {
  int i = blockIdx.x * blockDim.x + threadIdx.x;
  const int n_emb = 256 * 256, n_wx = 256 * 512, n_wh = 512 * 512, n_wd = 512 * 256;
  if (i < n_emb) { ws[EMB_OFF + i] = (f16)emb[i]; return; }
  i -= n_emb;
  if (i < n_wx) {  // WxT[n][k] = Wx[k][n], Wx is [256][512]
    int n = i / VOCABN, k = i % VOCABN;
    ws[WXT_OFF + i] = (f16)Wx[k * UNITS + n]; return;
  }
  i -= n_wx;
  if (i < n_wh) {  // WhT[n][k] = Wh[k][n], Wh is [512][512]
    int n = i / UNITS, k = i % UNITS;
    ws[WHT_OFF + i] = (f16)Wh[k * UNITS + n]; return;
  }
  i -= n_wh;
  if (i < n_wd) {  // WdT[n][k] = Wd[k][n], Wd is [512][256]
    int n = i / UNITS, k = i % UNITS;
    ws[WDT_OFF + i] = (f16)Wd[k * VOCABN + n];
  }
}

// A-fragment (16x32 f16, row-major source): lane half h reads K = k0+16h..+15
// from row (m0 + lane&15).  Two 16-byte loads.
__device__ __forceinline__ v16h load_a_frag(const f16* row, int k0, int kh) {
  union { v16h v; v8h h[2]; } u;
  u.h[0] = *(const v8h*)(row + k0 + 16 * kh);
  u.h[1] = *(const v8h*)(row + k0 + 16 * kh + 8);
  return u.v;
}
// B-fragment (32x16 f16) from N-major (transposed) weights: lane = column,
// half selects K block of 16 -> one contiguous 32-byte load.
__device__ __forceinline__ v16h load_b_frag(const f16* bt_col, int k0, int kh) {
  return *(const v16h*)(bt_col + k0 + 16 * kh);
}

// ---------------------------------------------------------------------------
// Phase B: xp[t][b][n] = (emb[inputs[m]] @ Wx)[m][n] + bias[n],  m = b*T + t
// One 16x16 tile per wave, K=256 -> 8 WMMAs.
// ---------------------------------------------------------------------------
__global__ void __launch_bounds__(256)
xp_gemm(const int* __restrict__ inputs, const f16* __restrict__ ws,
        const float* __restrict__ bias, f16* __restrict__ xp) {
  const f16* emb_h = ws + EMB_OFF;
  const f16* WxT   = ws + WXT_OFF;
  const int wave = threadIdx.x >> 5, lane = threadIdx.x & 31;
  const int kh = lane >> 4, l16 = lane & 15;
  const int tile = blockIdx.x * 8 + wave;        // 4096 Mtiles x 32 Ntiles
  const int mi = tile >> 5, ni = tile & 31;
  const int m0 = mi * 16, n0 = ni * 16;

  const f16* arow = emb_h + (size_t)inputs[m0 + l16] * VOCABN;  // gathered row
  const f16* bcol = WxT + (size_t)(n0 + l16) * VOCABN;
  v8f acc = {};
  #pragma unroll
  for (int k0 = 0; k0 < VOCABN; k0 += 32) {
    v16h a = load_a_frag(arow, k0, kh);
    v16h b = load_b_frag(bcol, k0, kh);
    acc = __builtin_amdgcn_wmma_f32_16x16x32_f16(false, a, false, b,
                                                 (short)0, acc, false, false);
  }
  const int n = n0 + l16;
  const float bn = bias[n];
  #pragma unroll
  for (int j = 0; j < 8; ++j) {                   // D: VGPR j -> M = j + 8*kh
    int m = m0 + 8 * kh + j;
    int b_ = m >> 10, t = m & 1023;               // m = b*1024 + t
    xp[((size_t)t * BATCH + b_) * UNITS + n] = (f16)(acc[j] + bn);
  }
}

// ---------------------------------------------------------------------------
// Phase C: sequential scan.  One workgroup, 16 waves (4 per SIMD -> room for
// ~256 VGPRs/wave).  h (64x512 f16 = 64KB) in LDS, XOR-swizzled in 8-f16
// granules.  Wave w owns two N-tiles:
//   tile  w      : B-fragments preloaded in 128 VGPRs (Wh is time-invariant)
//   tile  w + 16 : B-fragments streamed from L2 (16KB/wave/step)
// Each LDS A-fragment feeds two WMMAs.  128 WMMAs/wave/step.
// ---------------------------------------------------------------------------
__global__ void __launch_bounds__(512)
rnn_scan(const f16* __restrict__ ws, f16* __restrict__ hs,
         const f16* __restrict__ xp) {
  __shared__ f16 hbuf[BATCH * UNITS];             // exactly 64 KB
  const f16* WhT = ws + WHT_OFF;
  const int tid = threadIdx.x;
  const int wave = tid >> 5, lane = tid & 31;
  const int kh = lane >> 4, l16 = lane & 15;
  const int na0 = wave * 16;                      // register-resident N-tile
  const int nb0 = 256 + wave * 16;                // streamed N-tile
  const int rx = l16 & 7;                         // XOR key (rows mi*16+l16: mi*16 % 8 == 0)

  for (int i = tid; i < BATCH * UNITS; i += 512) hbuf[i] = (f16)0.f;

  // Preload loop-invariant Wh column strip for tile na0: 16 x v16h = 128 VGPRs.
  const f16* acol = WhT + (size_t)(na0 + l16) * UNITS + 16 * kh;
  const f16* bcol = WhT + (size_t)(nb0 + l16) * UNITS + 16 * kh;
  v16h bf[16];
  #pragma unroll
  for (int kk = 0; kk < 16; ++kk)
    bf[kk] = *(const v16h*)(acol + kk * 32);

  __syncthreads();

  for (int t = 0; t < TLEN; ++t) {
    v8f acc[8] = {};                              // [mi][tile a/b]
    #pragma unroll
    for (int kk = 0; kk < 16; ++kk) {
      const int c0 = kk * 4 + 2 * kh;             // 8-f16 granule index
      v16h b2 = *(const v16h*)(bcol + kk * 32);   // streamed tile, once per kk
      #pragma unroll
      for (int mi = 0; mi < 4; ++mi) {
        const f16* hrow = hbuf + (mi * 16 + l16) * UNITS;
        union { v16h v; v8h h[2]; } ua;
        ua.h[0] = *(const v8h*)(hrow + ((c0 ^ rx) << 3));
        ua.h[1] = *(const v8h*)(hrow + (((c0 + 1) ^ rx) << 3));
        acc[mi * 2 + 0] = __builtin_amdgcn_wmma_f32_16x16x32_f16(
            false, ua.v, false, bf[kk], (short)0, acc[mi * 2 + 0], false, false);
        acc[mi * 2 + 1] = __builtin_amdgcn_wmma_f32_16x16x32_f16(
            false, ua.v, false, b2, (short)0, acc[mi * 2 + 1], false, false);
      }
    }
    __syncthreads();                              // all reads of h done

    const f16* xpt = xp + (size_t)t * BATCH * UNITS;
    f16* hst = hs + (size_t)t * BATCH * UNITS;
    if (t + 1 < TLEN)                             // warm L2/WGP$ for next step
      __builtin_prefetch(xpt + (size_t)BATCH * UNITS + (wave * 4 + l16 / 4) * UNITS, 0, 3);
    #pragma unroll
    for (int mi = 0; mi < 4; ++mi) {
      #pragma unroll
      for (int s = 0; s < 2; ++s) {               // s=0: reg tile, s=1: streamed
        const int n = (s ? nb0 : na0) + l16;
        const int cg = n >> 3, nl = n & 7;
        #pragma unroll
        for (int r = 0; r < 8; ++r) {             // D: VGPR r -> M = r + 8*kh
          const int m = mi * 16 + 8 * kh + r;
          float v = tanhf(acc[mi * 2 + s][r] + (float)xpt[m * UNITS + n]);
          f16 hv = (f16)v;
          hbuf[m * UNITS + ((cg ^ (m & 7)) << 3) + nl] = hv;  // swizzled store
          hst[m * UNITS + n] = hv;
        }
      }
    }
    __syncthreads();                              // h visible for next step
  }
}

// ---------------------------------------------------------------------------
// Phase D: logits[m][n] = hs_row(m) @ WdT[n] + bd[n], f32 out.
// ---------------------------------------------------------------------------
__global__ void __launch_bounds__(256)
out_gemm(const f16* __restrict__ ws, const float* __restrict__ bd,
         float* __restrict__ out) {
  const f16* WdT = ws + WDT_OFF;
  const f16* hs  = ws + HS_OFF;
  const int wave = threadIdx.x >> 5, lane = threadIdx.x & 31;
  const int kh = lane >> 4, l16 = lane & 15;
  const int tile = blockIdx.x * 8 + wave;         // 4096 Mtiles x 16 Ntiles
  const int mi = tile >> 4, ni = tile & 15;
  const int m0 = mi * 16, n0 = ni * 16;

  const int m = m0 + l16;
  const int b_ = m >> 10, t = m & 1023;
  const f16* arow = hs + ((size_t)t * BATCH + b_) * UNITS;   // time-major hs
  const f16* bcol = WdT + (size_t)(n0 + l16) * UNITS;
  v8f acc = {};
  #pragma unroll 4
  for (int k0 = 0; k0 < UNITS; k0 += 32) {
    v16h a = load_a_frag(arow, k0, kh);
    v16h b = load_b_frag(bcol, k0, kh);
    acc = __builtin_amdgcn_wmma_f32_16x16x32_f16(false, a, false, b,
                                                 (short)0, acc, false, false);
  }
  const int n = n0 + l16;
  const float bn = bd[n];
  #pragma unroll
  for (int j = 0; j < 8; ++j) {
    int mm = m0 + 8 * kh + j;
    out[(size_t)mm * VOCABN + n] = acc[j] + bn;
  }
}

// ---------------------------------------------------------------------------
extern "C" void kernel_launch(void* const* d_in, const int* in_sizes, int n_in,
                              void* d_out, int out_size, void* d_ws, size_t ws_size,
                              hipStream_t stream) {
  const int*   inputs = (const int*)d_in[0];
  const float* emb    = (const float*)d_in[1];
  const float* Wx     = (const float*)d_in[2];
  const float* Wh     = (const float*)d_in[3];
  const float* bias   = (const float*)d_in[4];
  const float* Wd     = (const float*)d_in[5];
  const float* bd     = (const float*)d_in[6];
  float* out = (float*)d_out;
  f16*   ws  = (f16*)d_ws;

  // A: convert + transpose weights (589,824 elements)
  convert_weights<<<(589824 + 255) / 256, 256, 0, stream>>>(emb, Wx, Wh, Wd, ws);
  // B: xp GEMM, 131072 tiles / 8 waves per block
  xp_gemm<<<16384, 256, 0, stream>>>(inputs, ws, bias, ws + XP_OFF);
  // C: sequential scan, single persistent workgroup (16 waves)
  rnn_scan<<<1, 512, 0, stream>>>(ws, ws + HS_OFF, ws + XP_OFF);
  // D: output GEMM, 65536 tiles / 8 waves per block
  out_gemm<<<8192, 256, 0, stream>>>(ws, bd, out);
}
